// TemporalEncoding_78950088835528
// MI455X (gfx1250) — matrix operand order repaired and checked by reading, not compiled, not run on hardware
//
#include <hip/hip_runtime.h>
#include <math.h>

#define D_DIM  256
#define N_ROWS 100000
#define M_ROWS 50000

typedef __attribute__((ext_vector_type(16))) __bf16 v16bf;
typedef __attribute__((ext_vector_type(8)))  __bf16 v8bf;
typedef __attribute__((ext_vector_type(8)))  float  v8f;

__device__ __forceinline__ void split_bf16(float x, __bf16& h, __bf16& l) {
    h = (__bf16)x;
    l = (__bf16)(x - (float)h);
}

// Assemble a 16-bit A-matrix fragment from an LDS row (32 bf16 K values).
// Contiguous K order: lane<16 holds K = [k0,k0+16) with k0=0;
// lane>=16 uses k0=16. Element s = K k0+s (matches B fragments).
__device__ __forceinline__ v16bf lds_afrag(const __bf16* rowp, int k0) {
    v8bf x = *(const v8bf*)(rowp + k0);
    v8bf y = *(const v8bf*)(rowp + k0 + 8);
    return __builtin_shufflevector(x, y, 0,1,2,3,4,5,6,7,8,9,10,11,12,13,14,15);
}

__device__ __forceinline__ void load8(const float* p, float v[8]) {
    const float4* s = (const float4*)p;
    float4 u0 = s[0], u1 = s[1];
    v[0]=u0.x; v[1]=u0.y; v[2]=u0.z; v[3]=u0.w;
    v[4]=u1.x; v[5]=u1.y; v[6]=u1.z; v[7]=u1.w;
}

// ---------------------------------------------------------------------------
// Prep: split weights into bf16 hi/lo. weight_* are transposed to [n][k] so
// B fragments are contiguous; linear_w stays [j][k] (B = linear_w^T already).
// ---------------------------------------------------------------------------
__global__ __launch_bounds__(256) void prep_weights(
    const float* __restrict__ w_o, const float* __restrict__ w_p,
    const float* __restrict__ w_k, const float* __restrict__ lw,
    __bf16* __restrict__ wto_hi, __bf16* __restrict__ wto_lo,
    __bf16* __restrict__ wtp_hi, __bf16* __restrict__ wtp_lo,
    __bf16* __restrict__ wtk_hi, __bf16* __restrict__ wtk_lo,
    __bf16* __restrict__ lw_hi,  __bf16* __restrict__ lw_lo) {
    int i = blockIdx.x * 256 + threadIdx.x;
    if (i < D_DIM * 3 * D_DIM) {            // 256*768 linear_w elements
        __bf16 h, l;
        split_bf16(lw[i], h, l);
        lw_hi[i] = h; lw_lo[i] = l;
    }
    if (i < D_DIM * D_DIM) {                // 256*256 each weight, transpose
        int k = i >> 8, n = i & 255;
        int t = n * D_DIM + k;
        __bf16 h, l;
        split_bf16(w_o[i], h, l); wto_hi[t] = h; wto_lo[t] = l;
        split_bf16(w_p[i], h, l); wtp_hi[t] = h; wtp_lo[t] = l;
        split_bf16(w_k[i], h, l); wtk_hi[t] = h; wtk_lo[t] = l;
    }
}

// ---------------------------------------------------------------------------
// GEMM: out[N,256] = X[N,256] @ W[256,256], W given pre-transposed bf16 hi/lo.
// Block tile 64 rows x 256 cols, 8 waves (2M x 4N), wave tile 32x64.
// Double-buffered LDS A staging, one barrier per K-step; global loads for
// step k+1 issued before the WMMA phase of step k.
// ---------------------------------------------------------------------------
__global__ __launch_bounds__(256) void gemm_xw(
    const float* __restrict__ X,
    const __bf16* __restrict__ bt_hi, const __bf16* __restrict__ bt_lo,
    float* __restrict__ out, int nrows) {
    __shared__ alignas(16) __bf16 a_hi[2][64][32];
    __shared__ alignas(16) __bf16 a_lo[2][64][32];

    const int t    = threadIdx.x;
    const int lane = t & 31, wid = t >> 5;
    const int wm   = wid >> 2, wn = wid & 3;
    const int lrow = lane & 15, lgrp = lane >> 4;
    const int ka   = 16 * lgrp;          // A-frag K group
    const int kbo  = 16 * lgrp;          // B-frag K group
    const int row_base = blockIdx.x * 64;

    const int   mloc = t >> 2;
    const int   c0   = (t & 3) * 8;
    const int   grow = row_base + mloc;
    const bool  rv   = grow < nrows;
    const float* __restrict__ xrow = X + (size_t)grow * D_DIM;

    v8f acc[2][4] = {};

    float v[8];
    #pragma unroll
    for (int i = 0; i < 8; i++) v[i] = 0.f;
    if (rv) load8(xrow + c0, v);                     // prologue: tile kb=0

    for (int kb = 0; kb < D_DIM; kb += 32) {
        const int buf = (kb >> 5) & 1;
        // stage tile kb (held in v) into LDS as bf16 hi/lo
        #pragma unroll
        for (int i = 0; i < 8; i++) {
            __bf16 h, l; split_bf16(v[i], h, l);
            a_hi[buf][mloc][c0 + i] = h;
            a_lo[buf][mloc][c0 + i] = l;
        }
        __syncthreads();

        // issue global loads for tile kb+32 (consumed next iteration)
        if (rv && (kb + 32 < D_DIM)) load8(xrow + kb + 32 + c0, v);

        v16bf afh[2], afl[2];
        #pragma unroll
        for (int tm = 0; tm < 2; tm++) {
            int r = wm * 32 + tm * 16 + lrow;
            afh[tm] = lds_afrag(&a_hi[buf][r][0], ka);
            afl[tm] = lds_afrag(&a_lo[buf][r][0], ka);
        }
        #pragma unroll
        for (int tn = 0; tn < 4; tn++) {
            int nabs = wn * 64 + tn * 16 + lrow;
            v16bf bh = *(const v16bf*)(bt_hi + (size_t)nabs * D_DIM + kb + kbo);
            v16bf bl = *(const v16bf*)(bt_lo + (size_t)nabs * D_DIM + kb + kbo);
            #pragma unroll
            for (int tm = 0; tm < 2; tm++) {
                acc[tm][tn] = __builtin_amdgcn_wmma_f32_16x16x32_bf16(
                    false, afh[tm], false, bh, (short)0, acc[tm][tn], false, false);
                acc[tm][tn] = __builtin_amdgcn_wmma_f32_16x16x32_bf16(
                    false, afl[tm], false, bh, (short)0, acc[tm][tn], false, false);
                acc[tm][tn] = __builtin_amdgcn_wmma_f32_16x16x32_bf16(
                    false, afh[tm], false, bl, (short)0, acc[tm][tn], false, false);
            }
        }
        // no trailing barrier: double buffering makes one barrier/iter safe
    }

    // --- store: C layout lane<16 -> M=r, lane>=16 -> M=8+r, N=lane&15 ---
    #pragma unroll
    for (int tm = 0; tm < 2; tm++) {
        #pragma unroll
        for (int r = 0; r < 8; r++) {
            int ra = row_base + wm * 32 + tm * 16 + r + 8 * lgrp;
            if (ra < nrows) {
                #pragma unroll
                for (int tn = 0; tn < 4; tn++) {
                    int col = wn * 64 + tn * 16 + lrow;
                    out[(size_t)ra * D_DIM + col] = acc[tm][tn][r];
                }
            }
        }
    }
}

// ---------------------------------------------------------------------------
// Fused tail: out = leaky_relu(tanh(gather) @ linear_w^T + linear_b) + bias,
// scattered into trans_hp[ip] and trans_hk[ik]. K=768 in 3 source segments.
// Same double-buffered pipeline; tanh + bf16 split happen in the store phase.
// ---------------------------------------------------------------------------
__global__ __launch_bounds__(256) void fused_tail(
    const float* __restrict__ tho,
    float* __restrict__ thp, float* __restrict__ thk,
    const int* __restrict__ ip, const int* __restrict__ ik,
    const __bf16* __restrict__ lwh, const __bf16* __restrict__ lwl,
    const float* __restrict__ lb, const float* __restrict__ bias) {
    __shared__ alignas(16) __bf16 a_hi[2][64][32];
    __shared__ alignas(16) __bf16 a_lo[2][64][32];
    __shared__ int sip[64];
    __shared__ int sik[64];

    const int t    = threadIdx.x;
    const int lane = t & 31, wid = t >> 5;
    const int wm   = wid >> 2, wn = wid & 3;
    const int lrow = lane & 15, lgrp = lane >> 4;
    const int ka   = 16 * lgrp;
    const int kbo  = 16 * lgrp;
    const int row_base = blockIdx.x * 64;

    if (t < 64) {
        int m = row_base + t;
        sip[t] = (m < M_ROWS) ? ip[m] : 0;
        sik[t] = (m < M_ROWS) ? ik[m] : 0;
    }
    __syncthreads();

    const int  mloc = t >> 2;
    const int  c0   = (t & 3) * 8;
    const bool rv   = (row_base + mloc) < M_ROWS;
    const size_t rowp_off = (size_t)sip[mloc] * D_DIM;   // gather rows for this
    const size_t rowk_off = (size_t)sik[mloc] * D_DIM;   // thread's staging row

    v8f acc[2][4] = {};

    float v[8];
    #pragma unroll
    for (int i = 0; i < 8; i++) v[i] = 0.f;
    if (rv) load8(tho + rowp_off + c0, v);               // prologue: kb=0 (h_o seg)

    for (int kb = 0; kb < 3 * D_DIM; kb += 32) {
        const int buf = (kb >> 5) & 1;
        // stage tile kb: tanh + bf16 hi/lo split
        #pragma unroll
        for (int i = 0; i < 8; i++) {
            __bf16 h, l; split_bf16(tanhf(v[i]), h, l);
            a_hi[buf][mloc][c0 + i] = h;
            a_lo[buf][mloc][c0 + i] = l;
        }
        __syncthreads();

        // issue gather loads for tile kb+32
        const int kn = kb + 32;
        if (rv && kn < 3 * D_DIM) {
            const float* src; size_t roff; int kc;
            if (kn < D_DIM)          { src = tho; roff = rowp_off; kc = kn;            }
            else if (kn < 2 * D_DIM) { src = thp; roff = rowp_off; kc = kn - D_DIM;    }
            else                     { src = thk; roff = rowk_off; kc = kn - 2*D_DIM;  }
            load8(src + roff + kc + c0, v);
        }

        v16bf afh[2], afl[2];
        #pragma unroll
        for (int tm = 0; tm < 2; tm++) {
            int r = wm * 32 + tm * 16 + lrow;
            afh[tm] = lds_afrag(&a_hi[buf][r][0], ka);
            afl[tm] = lds_afrag(&a_lo[buf][r][0], ka);
        }
        #pragma unroll
        for (int tn = 0; tn < 4; tn++) {
            int nabs = wn * 64 + tn * 16 + lrow;
            v16bf bh = *(const v16bf*)(lwh + (size_t)nabs * (3*D_DIM) + kb + kbo);
            v16bf bl = *(const v16bf*)(lwl + (size_t)nabs * (3*D_DIM) + kb + kbo);
            #pragma unroll
            for (int tm = 0; tm < 2; tm++) {
                acc[tm][tn] = __builtin_amdgcn_wmma_f32_16x16x32_bf16(
                    false, afh[tm], false, bh, (short)0, acc[tm][tn], false, false);
                acc[tm][tn] = __builtin_amdgcn_wmma_f32_16x16x32_bf16(
                    false, afl[tm], false, bh, (short)0, acc[tm][tn], false, false);
                acc[tm][tn] = __builtin_amdgcn_wmma_f32_16x16x32_bf16(
                    false, afh[tm], false, bl, (short)0, acc[tm][tn], false, false);
            }
        }
    }

    // epilogue: + linear_b, leaky_relu(0.01), + bias, scatter to both outputs
    #pragma unroll
    for (int tm = 0; tm < 2; tm++) {
        #pragma unroll
        for (int r = 0; r < 8; r++) {
            int mrel = wm * 32 + tm * 16 + r + 8 * lgrp;
            int mabs = row_base + mrel;
            if (mabs < M_ROWS) {
                int rp = sip[mrel];
                int rk = sik[mrel];
                #pragma unroll
                for (int tn = 0; tn < 4; tn++) {
                    int j = wn * 64 + tn * 16 + lrow;
                    float val = acc[tm][tn][r] + lb[j];
                    val = val > 0.f ? val : 0.01f * val;
                    val += bias[j];
                    thp[(size_t)rp * D_DIM + j] = val;
                    thk[(size_t)rk * D_DIM + j] = val;
                }
            }
        }
    }
}

// ---------------------------------------------------------------------------
extern "C" void kernel_launch(void* const* d_in, const int* in_sizes, int n_in,
                              void* d_out, int out_size, void* d_ws, size_t ws_size,
                              hipStream_t stream) {
    const float* h_p  = (const float*)d_in[0];
    const float* h_k  = (const float*)d_in[1];
    const float* h_o  = (const float*)d_in[2];
    const int*   ip   = (const int*)  d_in[3];
    const int*   ik   = (const int*)  d_in[4];
    const float* w_o  = (const float*)d_in[5];
    const float* w_p  = (const float*)d_in[6];
    const float* w_k  = (const float*)d_in[7];
    const float* lw   = (const float*)d_in[8];
    const float* lb   = (const float*)d_in[9];
    const float* bias = (const float*)d_in[10];

    float* out_hp = (float*)d_out;
    float* out_hk = out_hp + (size_t)N_ROWS * D_DIM;

    // workspace layout
    char*  ws  = (char*)d_ws;
    float* tho = (float*)ws;            ws += (size_t)N_ROWS * D_DIM * sizeof(float);
    __bf16* wto_hi = (__bf16*)ws;       ws += (size_t)D_DIM * D_DIM * 2;
    __bf16* wto_lo = (__bf16*)ws;       ws += (size_t)D_DIM * D_DIM * 2;
    __bf16* wtp_hi = (__bf16*)ws;       ws += (size_t)D_DIM * D_DIM * 2;
    __bf16* wtp_lo = (__bf16*)ws;       ws += (size_t)D_DIM * D_DIM * 2;
    __bf16* wtk_hi = (__bf16*)ws;       ws += (size_t)D_DIM * D_DIM * 2;
    __bf16* wtk_lo = (__bf16*)ws;       ws += (size_t)D_DIM * D_DIM * 2;
    __bf16* lw_hi  = (__bf16*)ws;       ws += (size_t)D_DIM * 3 * D_DIM * 2;
    __bf16* lw_lo  = (__bf16*)ws;       ws += (size_t)D_DIM * 3 * D_DIM * 2;

    prep_weights<<<(D_DIM * 3 * D_DIM + 255) / 256, 256, 0, stream>>>(
        w_o, w_p, w_k, lw,
        wto_hi, wto_lo, wtp_hi, wtp_lo, wtk_hi, wtk_lo, lw_hi, lw_lo);

    const int gblocks = (N_ROWS + 63) / 64;
    gemm_xw<<<gblocks, 256, 0, stream>>>(h_o, wto_hi, wto_lo, tho,    N_ROWS);
    gemm_xw<<<gblocks, 256, 0, stream>>>(h_p, wtp_hi, wtp_lo, out_hp, N_ROWS);
    gemm_xw<<<gblocks, 256, 0, stream>>>(h_k, wtk_hi, wtk_lo, out_hk, N_ROWS);

    fused_tail<<<(M_ROWS + 63) / 64, 256, 0, stream>>>(
        tho, out_hp, out_hk, ip, ik, lw_hi, lw_lo, lb, bias);
}
